// ContrastiveGNN_37151467111037
// MI455X (gfx1250) — matrix operand order repaired and checked by reading, not compiled, not run on hardware
//
#include <hip/hip_runtime.h>
#include <hip/hip_bf16.h>

// ---------------------------------------------------------------------------
// ContrastiveGNN edge classifier for MI455X (gfx1250, wave32, WMMA bf16)
//
// Memory-bound workload (23.3 TB/s HBM): fuse the edge MLP so the [E,256]
// edge-feature matrix (512MB) never exists in HBM. Dense GEMMs run on
// v_wmma_f32_16x16x32_bf16. Swizzled weights are staged into LDS once per
// block via the CDNA5 async-to-LDS path (ASYNCcnt + s_wait_asynccnt) and
// reused by persistent waves across many tiles.
// ---------------------------------------------------------------------------

typedef __attribute__((ext_vector_type(16))) __bf16 v16bf;
typedef __attribute__((ext_vector_type(8)))  float  v8f;

#define GNN_N 100000
#define GNN_E 500000
#define GNN_D 128

static __device__ __forceinline__ __bf16 to_bf16(float f) { return (__bf16)f; }

// CDNA5 async copy: 16B global -> LDS, tracked by ASYNCcnt.
// Generic pointers to LDS carry the LDS byte offset in their low 32 bits.
static __device__ __forceinline__ void async_copy_b128(void* lds, const void* g) {
  unsigned       l  = (unsigned)(size_t)lds;
  unsigned long long ga = (unsigned long long)(size_t)g;
  asm volatile("global_load_async_to_lds_b128 %0, %1, off"
               :: "v"(l), "v"(ga) : "memory");
}
static __device__ __forceinline__ void wait_async_zero() {
  asm volatile("s_wait_asynccnt 0x0" ::: "memory");
}

// B fragment: pre-swizzled so lane's 16 bf16 elements are contiguous.
// layout index = ((kt*NT + nt)*32 + lane)*16 + e
static __device__ __forceinline__ v16bf load_b_frag(const __bf16* __restrict__ Bsw,
                                                    int NT, int kt, int nt, int lane) {
  const __bf16* p = Bsw + ((((size_t)(kt * NT + nt) << 5) + lane) << 4);
  v16bf b;
#pragma unroll
  for (int e = 0; e < 16; ++e) b[e] = p[e];
  return b;
}

// A fragment (16x32 bf16) from fp32 row-major source; p points at row base + k0.
// CDNA5 16-bit A layout: elems 0..7 -> K = kh+0..7 ; elems 8..15 -> K = 16+kh+0..7
// where kh = (lane>>4)*8.
static __device__ __forceinline__ v16bf load_a_frag_f32(const float* __restrict__ p, int kh) {
  v16bf a;
#pragma unroll
  for (int e = 0; e < 8; ++e) a[e] = to_bf16(p[kh + e]);
#pragma unroll
  for (int e = 0; e < 8; ++e) a[8 + e] = to_bf16(p[16 + kh + e]);
  return a;
}

// ---------------------------------------------------------------------------
// Weight swizzle: fp32 [K x N] row-major -> bf16 WMMA-B lane layout.
// element e on lane L of tile (kt,nt):  K = kt*32 + (L>>4)*8 + (e&7) + (e>>3)*16, N = nt*16 + (L&15)
// ---------------------------------------------------------------------------
__global__ void gnn_swizzle_w(const float* __restrict__ W, __bf16* __restrict__ dst,
                              int K, int N) {
  int idx = blockIdx.x * blockDim.x + threadIdx.x;
  if (idx >= K * N) return;
  int e  = idx & 15;
  int L  = (idx >> 4) & 31;
  int t  = idx >> 9;          // tile = kt*NT + nt
  int NT = N >> 4;
  int nt = t % NT;
  int kt = t / NT;
  int k = kt * 32 + ((L >> 4) << 3) + (e & 7) + ((e >> 3) << 4);
  int n = nt * 16 + (L & 15);
  dst[idx] = to_bf16(W[k * N + n]);
}

// ---------------------------------------------------------------------------
// Degree / normalization:  deg = 1 (self loop) + in-degree(col); dinv = rsqrt(deg)
// ---------------------------------------------------------------------------
__global__ void gnn_deg_init(float* __restrict__ deg, int n) {
  int i = blockIdx.x * blockDim.x + threadIdx.x;
  if (i < n) deg[i] = 1.0f;
}
__global__ void gnn_deg_add(const int* __restrict__ colI, float* __restrict__ deg, int E) {
  int i = blockIdx.x * blockDim.x + threadIdx.x;
  if (i < E) atomicAdd(&deg[colI[i]], 1.0f);
}
__global__ void gnn_deg_rsqrt(float* __restrict__ deg, int n) {
  int i = blockIdx.x * blockDim.x + threadIdx.x;
  if (i < n) deg[i] = rsqrtf(deg[i]);
}

// ---------------------------------------------------------------------------
// Dense GEMM:  C[M x (NT*16)] = A[M x (KT*32)] @ B  (bf16 WMMA, fp32 accum)
// B staged in LDS via async copy; persistent waves stride over 16-row tiles.
// ---------------------------------------------------------------------------
template <int KT, int NT>
__global__ __launch_bounds__(128)
void gnn_gemm_bf16(const float* __restrict__ A, const __bf16* __restrict__ Bsw,
                   float* __restrict__ C, int M, int lda) {
  __shared__ __bf16 Bl[KT * NT * 512];                // 512 bf16 per 16x32x16 tile
  const int tid = threadIdx.x;

  // async stage B into LDS (KT*NT*1024 bytes in 16B chunks)
  for (int i = tid; i < KT * NT * 64; i += 128)
    async_copy_b128((char*)Bl + i * 16, (const char*)Bsw + i * 16);
  wait_async_zero();
  __syncthreads();

  const int lane = tid & 31;
  const int wave = tid >> 5;
  const int kh   = (lane >> 4) << 3;                  // 0 or 8
  const int nTilesM = M >> 4;
  const int waveStride = gridDim.x * 4;

  for (int tileM = blockIdx.x * 4 + wave; tileM < nTilesM; tileM += waveStride) {
    const int m0 = tileM * 16;
    v8f acc[NT] = {};
    for (int kt = 0; kt < KT; ++kt) {
      const float* arow = A + (size_t)(m0 + (lane & 15)) * lda + kt * 32;
      v16bf a = load_a_frag_f32(arow, kh);
#pragma unroll
      for (int nt = 0; nt < NT; ++nt) {
        v16bf b = load_b_frag(Bl, NT, kt, nt, lane);
        acc[nt] = __builtin_amdgcn_wmma_f32_16x16x32_bf16(
            false, a, false, b, (short)0, acc[nt], false, false);
      }
    }
    // D layout: VGPR r -> row m0 + (lane>>4)*8 + r, col nt*16 + (lane&15)
    const int rbase = m0 + ((lane >> 4) << 3);
    const int col0  = lane & 15;
#pragma unroll
    for (int nt = 0; nt < NT; ++nt)
#pragma unroll
      for (int r = 0; r < 8; ++r)
        C[(size_t)(rbase + r) * (NT * 16) + nt * 16 + col0] = acc[nt][r];
  }
}

// ---------------------------------------------------------------------------
// GCN aggregation:  out[i] = dinv[i]^2 * hw[i] + bias, then
// scatter out[col] += dinv[row]*dinv[col] * hw[row] with f32 atomics.
// ---------------------------------------------------------------------------
__global__ void gnn_selfloop_init(const float* __restrict__ hw, const float* __restrict__ dinv,
                                  const float* __restrict__ bias, float* __restrict__ out, int n) {
  int idx = blockIdx.x * blockDim.x + threadIdx.x;
  if (idx >= n * GNN_D) return;
  int node = idx >> 7;
  int c    = idx & (GNN_D - 1);
  float di = dinv[node];
  out[idx] = di * di * hw[idx] + bias[c];
}

__global__ void gnn_scatter(const float* __restrict__ hw, const int* __restrict__ rowI,
                            const int* __restrict__ colI, const float* __restrict__ dinv,
                            float* __restrict__ out, int E) {
  int t = blockIdx.x * blockDim.x + threadIdx.x;     // E * 32 threads, 4 ch each
  int e = t >> 5;
  if (e >= E) return;
  int c4 = (t & 31) << 2;
  int r = rowI[e], c = colI[e];
  float norm = dinv[r] * dinv[c];
  const float* src = hw + (size_t)r * GNN_D + c4;
  float*       dst = out + (size_t)c * GNN_D + c4;
#pragma unroll
  for (int j = 0; j < 4; ++j) atomicAdd(&dst[j], norm * src[j]);
}

__global__ void gnn_relu(float* __restrict__ p, int n) {
  int i = blockIdx.x * blockDim.x + threadIdx.x;
  if (i < n) p[i] = fmaxf(p[i], 0.0f);
}

// ---------------------------------------------------------------------------
// Fused edge MLP, persistent waves, weights resident in LDS.
// Dynamic LDS layout (bytes):
//   [0,      65536)  We   swizzled bf16  (256x128)
//   [65536,  81920)  Wc1  swizzled bf16  (128x64)
//   [81920, 114688)  per-wave 16x128 f32 scratch (8KB x 4 waves)
// Per 16-edge tile:
//   L1: [16x256] gathered concat @ We -> relu      (64 WMMA)
//   transpose through LDS (D-layout -> A-layout)
//   L2: [16x128] @ Wc1 -> relu                     (16 WMMA)
//   L3: [16x64] @ Wc2 + softmax on lanes 0..15
// ---------------------------------------------------------------------------
__global__ __launch_bounds__(128)
void gnn_edge_mlp(const float* __restrict__ emb, const int* __restrict__ rowI,
                  const int* __restrict__ colI,
                  const __bf16* __restrict__ WeS, const float* __restrict__ be,
                  const __bf16* __restrict__ Wc1S, const float* __restrict__ bc1,
                  const float* __restrict__ Wc2, const float* __restrict__ bc2,
                  float* __restrict__ out, int E) {
  extern __shared__ char sm[];
  __bf16* WeL  = (__bf16*)sm;                         // 32768 bf16
  __bf16* Wc1L = (__bf16*)(sm + 65536);               //  8192 bf16
  const int tid  = threadIdx.x;
  const int lane = tid & 31;
  const int wave = tid >> 5;
  float* L = (float*)(sm + 81920) + wave * 2048;      // 16x128 f32, per wave

  // async stage weights into LDS: 4096 + 1024 16B chunks across 128 threads
  for (int i = tid; i < 4096; i += 128)
    async_copy_b128((char*)WeL + i * 16, (const char*)WeS + i * 16);
  for (int i = tid; i < 1024; i += 128)
    async_copy_b128((char*)Wc1L + i * 16, (const char*)Wc1S + i * 16);
  wait_async_zero();
  __syncthreads();

  const int myrow = lane & 15;
  const int kh    = (lane >> 4) << 3;                 // 0 or 8
  const int rb    = (lane >> 4) << 3;
  const int c0    = lane & 15;
  const int nTiles = E >> 4;
  const int waveStride = gridDim.x * 4;

  for (int tile = blockIdx.x * 4 + wave; tile < nTiles; tile += waveStride) {
    const int e0    = tile * 16;
    const int medge = e0 + myrow;
    const int r = rowI[medge];
    const int c = colI[medge];
    const float* embR = emb + (size_t)r * GNN_D;
    const float* embC = emb + (size_t)c * GNN_D;

    // ---- layer 1: K = 256 (8 kt), N = 128 (8 nt) ----
    v8f acc1[8] = {};
    for (int kt = 0; kt < 8; ++kt) {
      v16bf a;
#pragma unroll
      for (int half = 0; half < 2; ++half) {
        int k0 = kt * 32 + half * 16 + kh;            // 8-long run, never crosses 128
        const float* src = (k0 < GNN_D) ? (embR + k0) : (embC + (k0 - GNN_D));
#pragma unroll
        for (int e = 0; e < 8; ++e) a[half * 8 + e] = to_bf16(src[e]);
      }
#pragma unroll
      for (int nt = 0; nt < 8; ++nt) {
        v16bf b = load_b_frag(WeL, 8, kt, nt, lane);
        acc1[nt] = __builtin_amdgcn_wmma_f32_16x16x32_bf16(
            false, a, false, b, (short)0, acc1[nt], false, false);
      }
    }

    // bias + relu, stage [16 x 128] in per-wave LDS (D-layout -> row-major)
#pragma unroll
    for (int nt = 0; nt < 8; ++nt)
#pragma unroll
      for (int rr = 0; rr < 8; ++rr) {
        float v = acc1[nt][rr] + be[nt * 16 + c0];
        L[(rb + rr) * 128 + nt * 16 + c0] = fmaxf(v, 0.0f);
      }

    // ---- layer 2: K = 128 (4 kt), N = 64 (4 nt), A re-read from LDS ----
    v8f acc2[4] = {};
    for (int kt = 0; kt < 4; ++kt) {
      v16bf a;
#pragma unroll
      for (int half = 0; half < 2; ++half) {
        int k0 = kt * 32 + half * 16 + kh;
        const float* src = L + myrow * 128 + k0;
#pragma unroll
        for (int e = 0; e < 8; ++e) a[half * 8 + e] = to_bf16(src[e]);
      }
#pragma unroll
      for (int nt = 0; nt < 4; ++nt) {
        v16bf b = load_b_frag(Wc1L, 4, kt, nt, lane);
        acc2[nt] = __builtin_amdgcn_wmma_f32_16x16x32_bf16(
            false, a, false, b, (short)0, acc2[nt], false, false);
      }
    }

    // bias + relu -> hcls [16 x 64]; overlaps L (all L reads already issued,
    // per-wave LDS ops complete in order so this is safe).
    float* H = L;
#pragma unroll
    for (int nt = 0; nt < 4; ++nt)
#pragma unroll
      for (int rr = 0; rr < 8; ++rr) {
        float v = acc2[nt][rr] + bc1[nt * 16 + c0];
        H[(rb + rr) * 64 + nt * 16 + c0] = fmaxf(v, 0.0f);
      }

    // ---- layer 3 + softmax: lanes 0..15, one edge each ----
    if (lane < 16) {
      const float* hv = H + lane * 64;
      float l0 = bc2[0], l1 = bc2[1];
#pragma unroll 8
      for (int k = 0; k < 64; ++k) {
        float h = hv[k];
        l0 = fmaf(h, Wc2[k * 2 + 0], l0);
        l1 = fmaf(h, Wc2[k * 2 + 1], l1);
      }
      float m  = fmaxf(l0, l1);
      float z0 = __expf(l0 - m);
      float z1 = __expf(l1 - m);
      out[e0 + lane] = z1 / (z0 + z1);
    }
  }
}

// ---------------------------------------------------------------------------
// Host-side orchestration (graph-capture safe: only kernel launches on stream)
// ---------------------------------------------------------------------------
extern "C" void kernel_launch(void* const* d_in, const int* in_sizes, int n_in,
                              void* d_out, int out_size, void* d_ws, size_t ws_size,
                              hipStream_t stream) {
  (void)in_sizes; (void)n_in; (void)out_size; (void)ws_size;

  const float* x   = (const float*)d_in[0];
  const int*   ei  = (const int*)  d_in[1];
  const float* W1  = (const float*)d_in[2];
  const float* b1  = (const float*)d_in[3];
  const float* W2  = (const float*)d_in[4];
  const float* b2  = (const float*)d_in[5];
  const float* We  = (const float*)d_in[6];
  const float* be  = (const float*)d_in[7];
  const float* Wc1 = (const float*)d_in[8];
  const float* bc1 = (const float*)d_in[9];
  const float* Wc2 = (const float*)d_in[10];
  const float* bc2 = (const float*)d_in[11];
  const int* rowI = ei;
  const int* colI = ei + GNN_E;

  char* ws = (char*)d_ws;
  size_t off = 0;
  auto take = [&](size_t bytes) -> char* {
    char* p = ws + off;
    off = (off + bytes + 255) & ~(size_t)255;
    return p;
  };
  float*  dinv = (float*) take((size_t)GNN_N * 4);
  __bf16* W1s  = (__bf16*)take(128 * 128 * 2);
  __bf16* W2s  = (__bf16*)take(128 * 128 * 2);
  __bf16* WeS  = (__bf16*)take(256 * 128 * 2);
  __bf16* Wc1S = (__bf16*)take(128 * 64 * 2);
  float*  bufA = (float*) take((size_t)GNN_N * GNN_D * 4);   // hw scratch
  float*  bufB = (float*) take((size_t)GNN_N * GNN_D * 4);   // aggregated

  auto cdiv = [](long long a, long long b) { return (int)((a + b - 1) / b); };

  // weight swizzle to bf16 WMMA-B layout
  gnn_swizzle_w<<<cdiv(128 * 128, 256), 256, 0, stream>>>(W1, W1s, 128, 128);
  gnn_swizzle_w<<<cdiv(128 * 128, 256), 256, 0, stream>>>(W2, W2s, 128, 128);
  gnn_swizzle_w<<<cdiv(256 * 128, 256), 256, 0, stream>>>(We, WeS, 256, 128);
  gnn_swizzle_w<<<cdiv(128 * 64, 256), 256, 0, stream>>>(Wc1, Wc1S, 128, 64);

  // symmetric normalization coefficients
  gnn_deg_init <<<cdiv(GNN_N, 256), 256, 0, stream>>>(dinv, GNN_N);
  gnn_deg_add  <<<cdiv(GNN_E, 256), 256, 0, stream>>>(colI, dinv, GNN_E);
  gnn_deg_rsqrt<<<cdiv(GNN_N, 256), 256, 0, stream>>>(dinv, GNN_N);

  const int gemmBlocks = 512;                          // persistent waves
  const long long nElems = (long long)GNN_N * GNN_D;

  // GCN conv 1: h = relu(norm-agg(x@W1) + b1)
  gnn_gemm_bf16<4, 8><<<gemmBlocks, 128, 0, stream>>>(x, W1s, bufA, GNN_N, GNN_D);
  gnn_selfloop_init<<<cdiv(nElems, 256), 256, 0, stream>>>(bufA, dinv, b1, bufB, GNN_N);
  gnn_scatter<<<cdiv((long long)GNN_E * 32, 256), 256, 0, stream>>>(bufA, rowI, colI, dinv, bufB, GNN_E);
  gnn_relu<<<cdiv(nElems, 256), 256, 0, stream>>>(bufB, (int)nElems);

  // GCN conv 2: node_emb = norm-agg(h@W2) + b2
  gnn_gemm_bf16<4, 8><<<gemmBlocks, 128, 0, stream>>>(bufB, W2s, bufA, GNN_N, GNN_D);
  gnn_selfloop_init<<<cdiv(nElems, 256), 256, 0, stream>>>(bufA, dinv, b2, bufB, GNN_N);
  gnn_scatter<<<cdiv((long long)GNN_E * 32, 256), 256, 0, stream>>>(bufA, rowI, colI, dinv, bufB, GNN_E);

  // fused edge MLP + softmax -> probs[:,1] (112KB dynamic LDS per block)
  gnn_edge_mlp<<<1024, 128, 114688, stream>>>(
      bufB, rowI, colI, WeS, be, Wc1S, bc1, Wc2, bc2, (float*)d_out, GNN_E);
}
